// DeformableDownBlock_31026843746312
// MI455X (gfx1250) — matrix operand order repaired and checked by reading, not compiled, not run on hardware
//
#include <hip/hip_runtime.h>
#include <cstdint>

// ---------------------------------------------------------------------------
// DeformableDownBlock on MI455X (gfx1250, wave32, WMMA)
//
//  * Both the main einsum (M=128) AND the offset/modulation convs (M=32,
//    18 off + 9 mod rows) run as WMMA GEMMs (v_wmma_f32_16x16x32_f16,
//    f32 accumulate) with weights pre-swizzled once per layer into the
//    per-lane A-operand layout (L2-resident; loaded straight to VGPRs).
//  * Bilinear sampling / im2col fused into B-operand LDS staging
//    (ping-pong, one barrier per K-step); patch matrices never touch HBM;
//    gathers are L2-resident (x/h1 = 16.8-33.5 MB << 192 MB L2).
//  * Branchless tap handling: clamped-index loads + select-zeroed weights
//    (no exec-mask branches in the staging inner loop); per-element /9, /3
//    replaced by incremental kx/ky/c tracking (consecutive kg per thread).
//  * BN+ReLU of layer 1 folded into layer-2 reads (monotone per-channel);
//    saves a full 67 MB HBM read+write pass.
// ---------------------------------------------------------------------------

typedef __attribute__((ext_vector_type(16))) _Float16 v16h;
typedef __attribute__((ext_vector_type(8)))  float    v8f;

#define TB     256    // threads per block (8 wave32)
#define DT     128    // spatial positions per GEMM block
#define OCH    128    // output channels (both layers)
#define NSPLIT 16     // bn_stats split factor

// A-operand order packed by this code (16-bit A 16x32; contiguous k, not the ISA table order):
//   M = (lane & 15), K = j + ((lane>=16)?16:0)
// B-operand order staged here (32x16, same k order as A): N = lane&15, K = j + ((lane>=16)?16:0)
// C/D map: M = r + 8*(lane>=16), N = lane&15

// ---------------- main-GEMM weight pre-swizzle (M=128 -> 8 subtiles) -------
__global__ void pack_weights_kernel(const float* __restrict__ w,
                                    _Float16* __restrict__ wswz,
                                    int KTOT, int total)
{
    int idx = blockIdx.x * TB + threadIdx.x;
    if (idx >= total) return;
    int j    = idx & 15;
    int lane = (idx >> 4) & 31;
    int sub  = (idx >> 9) & 7;
    int ks   = idx >> 12;
    int M    = (sub << 4) + (lane & 15);
    int K    = j + ((lane >= 16) ? 16 : 0);
    wswz[idx] = (_Float16)w[(size_t)M * KTOT + (ks << 5) + K];
}

// ---------------- offmod-GEMM weight pre-swizzle (M=32 -> 2 subtiles) ------
// rows 0..17 = w_off, 18..26 = w_mod, 27..31 = zero pad
__global__ void pack_offmod_weights_kernel(const float* __restrict__ w_off,
                                           const float* __restrict__ w_mod,
                                           _Float16* __restrict__ wz,
                                           int KTOT, int total)
{
    int idx = blockIdx.x * TB + threadIdx.x;
    if (idx >= total) return;
    int j    = idx & 15;
    int lane = (idx >> 4) & 31;
    int sub  = (idx >> 9) & 1;
    int ks   = idx >> 10;
    int M    = (sub << 4) + (lane & 15);
    int K    = j + ((lane >= 16) ? 16 : 0);
    int col  = (ks << 5) + K;
    float v  = 0.f;
    if (M < 18)      v = w_off[(size_t)M * KTOT + col];
    else if (M < 27) v = w_mod[(size_t)(M - 18) * KTOT + col];
    wz[idx] = (_Float16)v;
}

// ---------------- branchless im2col tap (optional fused BN+ReLU) -----------
template <bool NORM>
__device__ __forceinline__ float tap(const float* __restrict__ xc,
                                     int H, int W, int y, int x,
                                     float m, float is)
{
    bool ok = ((unsigned)y < (unsigned)H) & ((unsigned)x < (unsigned)W);
    int yc = min(max(y, 0), H - 1);
    int xcl = min(max(x, 0), W - 1);
    float v = xc[yc * W + xcl];
    if (NORM) v = fmaxf((v - m) * is, 0.f);
    return ok ? v : 0.f;
}

// ---------------- branchless bilinear (optional fused BN+ReLU per tap) -----
template <bool NORM>
__device__ __forceinline__ float bilinear(const float* __restrict__ xc,
                                          int H, int W, float py, float px,
                                          float m, float is)
{
    float y0f = floorf(py), x0f = floorf(px);
    int y0 = (int)y0f, x0 = (int)x0f;
    float wy = py - y0f, wx = px - x0f;
    int y1 = y0 + 1, x1 = x0 + 1;
    bool vy0 = (unsigned)y0 < (unsigned)H, vy1 = (unsigned)y1 < (unsigned)H;
    bool vx0 = (unsigned)x0 < (unsigned)W, vx1 = (unsigned)x1 < (unsigned)W;
    int y0c = min(max(y0, 0), H - 1), y1c = min(max(y1, 0), H - 1);
    int x0c = min(max(x0, 0), W - 1), x1c = min(max(x1, 0), W - 1);
    float v00 = xc[y0c * W + x0c];
    float v01 = xc[y0c * W + x1c];
    float v10 = xc[y1c * W + x0c];
    float v11 = xc[y1c * W + x1c];
    if (NORM) {
        v00 = fmaxf((v00 - m) * is, 0.f);
        v01 = fmaxf((v01 - m) * is, 0.f);
        v10 = fmaxf((v10 - m) * is, 0.f);
        v11 = fmaxf((v11 - m) * is, 0.f);
    }
    float f00 = (vy0 & vx0) ? (1.f - wy) * (1.f - wx) : 0.f;
    float f01 = (vy0 & vx1) ? (1.f - wy) * wx         : 0.f;
    float f10 = (vy1 & vx0) ? wy * (1.f - wx)         : 0.f;
    float f11 = (vy1 & vx1) ? wy * wx                 : 0.f;
    return v00 * f00 + v01 * f01 + v10 * f10 + v11 * f11;
}

// ---------------- offsets + modulation convs as WMMA GEMM ------------------
// grid: (HW/DT, B). M=32 (2 subtiles), N=DT=128, K=Cin*9.
// 8 waves: wave w owns N-chunk w (16 cols) for both M subtiles.
template <bool NORM>
__global__ void __launch_bounds__(TB)
offmod_wmma_kernel(const float* __restrict__ x,       // (B,Cin,H,W) raw
                   const _Float16* __restrict__ wz,   // pre-swizzled A (32 x K)
                   const float* __restrict__ b_off,
                   const float* __restrict__ b_mod,
                   const float* __restrict__ mean,
                   const float* __restrict__ istd,
                   float* __restrict__ off,           // (B,18,H,W)
                   float* __restrict__ mod,           // (B,9,H,W) = 2*sigmoid
                   int Cin, int H, int W)
{
    const int t    = threadIdx.x;
    const int wave = t >> 5;
    const int lane = t & 31;
    const int b    = blockIdx.y;
    const int HW   = H * W;
    const int tile_base = blockIdx.x * DT;
    const int nK   = (Cin * 9) >> 5;

    __shared__ __align__(32) _Float16 ldsB[2][32 * DT];   // 2 x 8 KB

    // per-thread staging constants: fixed N, 16 consecutive kg values
    const int flat0 = t * 16;
    const int nsub0 = flat0 >> 9;
    const int ln0   = (flat0 & 511) >> 4;
    const int Nn    = (nsub0 << 4) + (ln0 & 15);
    const int Kb    = (ln0 >= 16) ? 16 : 0;
    const int hw0   = tile_base + Nn;
    const int hh0   = hw0 / W, ww0 = hw0 - hh0 * W;

    const v8f vzero = {0.f, 0.f, 0.f, 0.f, 0.f, 0.f, 0.f, 0.f};
    v8f acc0 = vzero, acc1 = vzero;

    const float* xb = x + (size_t)b * Cin * HW;

    for (int ks = 0; ks < nK; ++ks) {
        // ---- stage B: im2col (32 x 128) f16
        _Float16* dst = &ldsB[ks & 1][flat0];
        int kg = (ks << 5) + Kb;
        int c  = kg / 9;
        int kk = kg - c * 9;
        int ky = kk / 3;
        int kx = kk - ky * 3;
#pragma unroll
        for (int i = 0; i < 16; ++i) {
            float m  = NORM ? mean[c] : 0.f;
            float is = NORM ? istd[c] : 1.f;
            dst[i] = (_Float16)tap<NORM>(xb + (size_t)c * HW, H, W,
                                         hh0 + ky - 1, ww0 + kx - 1, m, is);
            if (++kx == 3) { kx = 0; if (++ky == 3) { ky = 0; ++c; } }
        }
        __syncthreads();   // single barrier per K-step (ping-pong)

        v16h a0 = *(const v16h*)(wz + ((size_t)(2 * ks    ) << 9) + (lane << 4));
        v16h a1 = *(const v16h*)(wz + ((size_t)(2 * ks + 1) << 9) + (lane << 4));
        if (ks + 1 < nK)
            __builtin_prefetch(wz + ((size_t)(2 * ks + 2) << 9) + (lane << 4), 0, 0);

        v16h bm = *(const v16h*)(&ldsB[ks & 1][0] + (wave << 9) + (lane << 4));
        acc0 = __builtin_amdgcn_wmma_f32_16x16x32_f16(false, a0, false, bm,
                                                      (short)0, acc0, false, false);
        acc1 = __builtin_amdgcn_wmma_f32_16x16x32_f16(false, a1, false, bm,
                                                      (short)0, acc1, false, false);
    }

    // ---- write back rows 0..17 -> off (+bias), 18..26 -> mod (2*sigmoid)
    const int hw    = tile_base + (wave << 4) + (lane & 15);
    const int obase = (lane >= 16) ? 8 : 0;
#pragma unroll
    for (int r = 0; r < 8; ++r) {
        int o = obase + r;                         // 0..15: always offset rows
        off[((size_t)(b * 18 + o)) * HW + hw] = acc0[r] + b_off[o];
    }
#pragma unroll
    for (int r = 0; r < 8; ++r) {
        int o = 16 + obase + r;                    // 16..31
        float v = acc1[r];
        if (o < 18) {
            off[((size_t)(b * 18 + o)) * HW + hw] = v + b_off[o];
        } else if (o < 27) {
            mod[((size_t)(b * 9 + (o - 18))) * HW + hw] =
                2.0f / (1.0f + __expf(-(v + b_mod[o - 18])));
        }
    }
}

// ---------------- fused deform-sample + WMMA GEMM --------------------------
// grid: (HW/DT, B). M=128 (wave = M-subtile), N=DT=128 (8 WMMA per wave/step).
template <bool NORM>
__global__ void __launch_bounds__(TB)
deform_wmma_kernel(const float* __restrict__ xsrc,    // (B,Cin,H,W) raw
                   const float* __restrict__ off,     // (B,18,H,W)
                   const float* __restrict__ mod,     // (B,9,H,W)
                   const _Float16* __restrict__ wswz, // pre-swizzled A (128 x K)
                   const float* __restrict__ mean,
                   const float* __restrict__ istd,
                   float* __restrict__ out,           // (B,128,H,W)
                   int Cin, int H, int W)
{
    const int t    = threadIdx.x;
    const int wave = t >> 5;
    const int lane = t & 31;
    const int b    = blockIdx.y;
    const int HW   = H * W;
    const int tile_base = blockIdx.x * DT;
    const int nK   = (Cin * 9) >> 5;

    __shared__ __align__(32) _Float16 ldsB[2][32 * DT];  // 2 x 8 KB
    __shared__ float s_dy[9 * DT];
    __shared__ float s_dx[9 * DT];
    __shared__ float s_m [9 * DT];

    // cache per-tile offset/modulation fields (reused Cin times in K loop)
    for (int i = t; i < 9 * DT; i += TB) {
        int kk = i / DT, n = i - kk * DT;
        int hw = tile_base + n;
        s_dy[i] = off[((size_t)(b * 18 + kk * 2    )) * HW + hw];
        s_dx[i] = off[((size_t)(b * 18 + kk * 2 + 1)) * HW + hw];
        s_m [i] = mod[((size_t)(b * 9  + kk        )) * HW + hw];
    }
    __syncthreads();

    // per-thread staging constants: fixed N, 16 consecutive kg values
    const int flat0 = t * 16;
    const int nsub0 = flat0 >> 9;
    const int ln0   = (flat0 & 511) >> 4;
    const int Nn    = (nsub0 << 4) + (ln0 & 15);
    const int Kb    = (ln0 >= 16) ? 16 : 0;
    const int hw0   = tile_base + Nn;
    const int hh0   = hw0 / W, ww0 = hw0 - hh0 * W;

    const v8f vzero = {0.f, 0.f, 0.f, 0.f, 0.f, 0.f, 0.f, 0.f};
    v8f acc[8];
#pragma unroll
    for (int i = 0; i < 8; ++i) acc[i] = vzero;

    const float* xb = xsrc + (size_t)b * Cin * HW;

    for (int ks = 0; ks < nK; ++ks) {
        // ---- stage B: fused bilinear sampling -> (32 x 128) f16 in LDS
        _Float16* dst = &ldsB[ks & 1][flat0];
        int kg = (ks << 5) + Kb;
        int c  = kg / 9;
        int kk = kg - c * 9;
        int ky = kk / 3;
        int kx = kk - ky * 3;
#pragma unroll
        for (int i = 0; i < 16; ++i) {
            int si = kk * DT + Nn;
            float py = (float)(hh0 + ky - 1) + s_dy[si];
            float px = (float)(ww0 + kx - 1) + s_dx[si];
            float m  = NORM ? mean[c] : 0.f;
            float is = NORM ? istd[c] : 1.f;
            float v  = bilinear<NORM>(xb + (size_t)c * HW, H, W, py, px, m, is);
            dst[i] = (_Float16)(v * s_m[si]);
            if (++kk == 9) kk = 0;
            if (++kx == 3) { kx = 0; if (++ky == 3) { ky = 0; ++c; } }
        }
        __syncthreads();   // single barrier per K-step (ping-pong)

        // ---- A straight from pre-swizzled global (L2-resident)
        v16h a = *(const v16h*)(wswz + ((size_t)ks << 12) + (wave << 9) + (lane << 4));
        if (ks + 1 < nK)
            __builtin_prefetch(wswz + ((size_t)(ks + 1) << 12) + (wave << 9) + (lane << 4), 0, 0);

        const _Float16* bbuf = ldsB[ks & 1];
#pragma unroll
        for (int nsub = 0; nsub < 8; ++nsub) {
            v16h bm = *(const v16h*)(bbuf + (nsub << 9) + (lane << 4));
            acc[nsub] = __builtin_amdgcn_wmma_f32_16x16x32_f16(
                false, a, false, bm, (short)0, acc[nsub], false, false);
        }
    }

    // ---- write back: C/D layout M = r + 8*(lane>=16), N = lane&15
    const int obase = (wave << 4) + ((lane >= 16) ? 8 : 0);
#pragma unroll
    for (int nsub = 0; nsub < 8; ++nsub) {
        int hw = tile_base + (nsub << 4) + (lane & 15);
#pragma unroll
        for (int r = 0; r < 8; ++r) {
            out[((size_t)(b * OCH + obase + r)) * HW + hw] = acc[nsub][r];
        }
    }
}

// ---------------- batch-norm statistics: stage 1 (partial sums) ------------
__global__ void bn_stats_kernel(const float* __restrict__ h,
                                float* __restrict__ psum,
                                float* __restrict__ psum2,
                                int C, int HW, int B)
{
    int c = blockIdx.x, s = blockIdx.y, t = threadIdx.x;
    int N = B * HW;
    int chunk = N / NSPLIT;
    int start = s * chunk;
    float sum = 0.f, sq = 0.f;
    for (int idx = start + t; idx < start + chunk; idx += TB) {
        int b  = idx / HW;
        int hw = idx - b * HW;
        float v = h[((size_t)(b * C + c)) * HW + hw];
        sum += v; sq += v * v;
    }
    __shared__ float rs[TB], rq[TB];
    rs[t] = sum; rq[t] = sq;
    __syncthreads();
    for (int st = TB / 2; st > 0; st >>= 1) {
        if (t < st) { rs[t] += rs[t + st]; rq[t] += rq[t + st]; }
        __syncthreads();
    }
    if (t == 0) {
        psum [c * NSPLIT + s] = rs[0];
        psum2[c * NSPLIT + s] = rq[0];
    }
}

// ---------------- batch-norm statistics: stage 2 (finalize) ----------------
__global__ void bn_finalize_kernel(const float* __restrict__ psum,
                                   const float* __restrict__ psum2,
                                   float* __restrict__ mean,
                                   float* __restrict__ istd,
                                   int N)
{
    int c = threadIdx.x;   // one thread per channel (blockDim = OCH)
    float s = 0.f, q = 0.f;
#pragma unroll
    for (int i = 0; i < NSPLIT; ++i) {
        s += psum [c * NSPLIT + i];
        q += psum2[c * NSPLIT + i];
    }
    float m = s / (float)N;
    mean[c] = m;
    istd[c] = rsqrtf(q / (float)N - m * m + 1e-5f);
}

// ---------------- BN + ReLU + 2x2 maxpool (reads raw h2) -------------------
__global__ void bn_pool_kernel(const float* __restrict__ h,
                               const float* __restrict__ mean,
                               const float* __restrict__ istd,
                               float* __restrict__ out,
                               int C, int H, int W, long total)
{
    long idx = (long)blockIdx.x * TB + threadIdx.x;
    if (idx >= total) return;
    int Ho = H >> 1, Wo = W >> 1;
    int ow = (int)(idx % Wo);  long r = idx / Wo;
    int oh = (int)(r % Ho);    r /= Ho;
    int c  = (int)(r % C);
    int b  = (int)(r / C);
    float m = mean[c], is = istd[c];
    const float* base = h + ((size_t)(b * C + c)) * H * W + (size_t)(2 * oh) * W + 2 * ow;
    float v0 = (base[0]     - m) * is;
    float v1 = (base[1]     - m) * is;
    float v2 = (base[W]     - m) * is;
    float v3 = (base[W + 1] - m) * is;
    v0 = v0 > 0.f ? v0 : 0.f;  v1 = v1 > 0.f ? v1 : 0.f;
    v2 = v2 > 0.f ? v2 : 0.f;  v3 = v3 > 0.f ? v3 : 0.f;
    float mx = v0 > v1 ? v0 : v1;
    mx = mx > v2 ? mx : v2;
    mx = mx > v3 ? mx : v3;
    out[idx] = mx;
}

// ---------------------------------------------------------------------------
extern "C" void kernel_launch(void* const* d_in, const int* in_sizes, int n_in,
                              void* d_out, int out_size, void* d_ws, size_t ws_size,
                              hipStream_t stream)
{
    (void)in_sizes; (void)n_in; (void)out_size; (void)ws_size;

    const float* x      = (const float*)d_in[0];
    const float* w_off1 = (const float*)d_in[1];
    const float* b_off1 = (const float*)d_in[2];
    const float* w_mod1 = (const float*)d_in[3];
    const float* b_mod1 = (const float*)d_in[4];
    const float* w1     = (const float*)d_in[5];
    const float* w_off2 = (const float*)d_in[6];
    const float* b_off2 = (const float*)d_in[7];
    const float* w_mod2 = (const float*)d_in[8];
    const float* b_mod2 = (const float*)d_in[9];
    const float* w2     = (const float*)d_in[10];
    float* out = (float*)d_out;

    const int B = 4, Cin = 64, H = 128, W = 128;
    const int HW = H * W;
    const int KTOT1 = Cin * 9, KTOT2 = OCH * 9;

    // workspace carve-up (floats): ~74.7 MB total
    float* ws    = (float*)d_ws;
    float* off   = ws;                                   // B*18*HW
    float* modb  = off   + (size_t)B * 18 * HW;          // B*9*HW
    float* h1    = modb  + (size_t)B * 9 * HW;           // B*128*HW (raw)
    float* h2    = h1    + (size_t)B * OCH * HW;         // B*128*HW (raw)
    float* mean1 = h2    + (size_t)B * OCH * HW;         // 128
    float* istd1 = mean1 + OCH;                          // 128
    float* mean2 = istd1 + OCH;                          // 128
    float* istd2 = mean2 + OCH;                          // 128
    float* psum  = istd2 + OCH;                          // 128*NSPLIT
    float* psum2 = psum  + OCH * NSPLIT;                 // 128*NSPLIT
    _Float16* wswz   = (_Float16*)(psum2 + OCH * NSPLIT);  // 128*KTOT2 halves max
    _Float16* wswzOM = wswz + (size_t)OCH * KTOT2;         // 32*KTOT2 halves max

    dim3 gemm_grid(HW / DT, B);
    dim3 stats_grid(OCH, NSPLIT);

    // ---------------- layer 1 (raw x, no BN folding) ----------------
    pack_offmod_weights_kernel<<<(32 * KTOT1 + TB - 1) / TB, TB, 0, stream>>>(
        w_off1, w_mod1, wswzOM, KTOT1, 32 * KTOT1);
    pack_weights_kernel<<<(OCH * KTOT1 + TB - 1) / TB, TB, 0, stream>>>(
        w1, wswz, KTOT1, OCH * KTOT1);

    offmod_wmma_kernel<false><<<gemm_grid, TB, 0, stream>>>(
        x, wswzOM, b_off1, b_mod1, nullptr, nullptr, off, modb, Cin, H, W);

    deform_wmma_kernel<false><<<gemm_grid, TB, 0, stream>>>(
        x, off, modb, wswz, nullptr, nullptr, h1, Cin, H, W);

    bn_stats_kernel<<<stats_grid, TB, 0, stream>>>(h1, psum, psum2, OCH, HW, B);
    bn_finalize_kernel<<<1, OCH, 0, stream>>>(psum, psum2, mean1, istd1, B * HW);

    // ---------------- layer 2 (BN+ReLU folded into reads of raw h1) --------
    pack_offmod_weights_kernel<<<(32 * KTOT2 + TB - 1) / TB, TB, 0, stream>>>(
        w_off2, w_mod2, wswzOM, KTOT2, 32 * KTOT2);
    pack_weights_kernel<<<(OCH * KTOT2 + TB - 1) / TB, TB, 0, stream>>>(
        w2, wswz, KTOT2, OCH * KTOT2);

    offmod_wmma_kernel<true><<<gemm_grid, TB, 0, stream>>>(
        h1, wswzOM, b_off2, b_mod2, mean1, istd1, off, modb, OCH, H, W);

    deform_wmma_kernel<true><<<gemm_grid, TB, 0, stream>>>(
        h1, off, modb, wswz, mean1, istd1, h2, OCH, H, W);

    bn_stats_kernel<<<stats_grid, TB, 0, stream>>>(h2, psum, psum2, OCH, HW, B);
    bn_finalize_kernel<<<1, OCH, 0, stream>>>(psum, psum2, mean2, istd2, B * HW);

    long totp = (long)B * OCH * (H / 2) * (W / 2);
    bn_pool_kernel<<<(unsigned)((totp + TB - 1) / TB), TB, 0, stream>>>(
        h2, mean2, istd2, out, OCH, H, W, totp);
}
